// HeteGNN_61744449847991
// MI455X (gfx1250) — matrix operand order repaired and checked by reading, not compiled, not run on hardware
//
#include <hip/hip_runtime.h>
#include <hip/hip_bf16.h>
#include <stdint.h>

#define NNODES 100000
#define NEDGES 600000
#define DIM 128
#define NLAYER 3
#define NGRAPH 64

// LDS row padding: 128 bf16 = 64 dwords; pad to 65 dwords (130 u16) so the
// 16 row-parallel fragment loads hit distinct banks.
#define LDU16 130
#define LDU32 65

typedef __attribute__((ext_vector_type(16))) __bf16 v16bf;
typedef __attribute__((ext_vector_type(8)))  float  v8f;

union BF16x16 { uint32_t u[8]; v16bf v; };

__device__ __forceinline__ uint16_t f2bf(float f) {
  uint32_t u = __float_as_uint(f);
  u += 0x7fffu + ((u >> 16) & 1u);          // round-to-nearest-even
  return (uint16_t)(u >> 16);
}
__device__ __forceinline__ uint32_t flipf(float f) {   // order-preserving float->uint
  uint32_t u = __float_as_uint(f);
  return (u & 0x80000000u) ? ~u : (u | 0x80000000u);
}
__device__ __forceinline__ float unflipf(uint32_t u) {
  return (u & 0x80000000u) ? __uint_as_float(u ^ 0x80000000u) : __uint_as_float(~u);
}

// ---------------- node input embedding ----------------
__global__ void k_embed(const int* __restrict__ x, const float* __restrict__ temb,
                        const float* __restrict__ oemb, float* __restrict__ h) {
  int t = blockIdx.x * blockDim.x + threadIdx.x;
  if (t >= NNODES * DIM) return;
  int n = t >> 7, c = t & 127;
  h[t] = temb[x[2 * n] * DIM + c] + oemb[x[2 * n + 1] * DIM + c];
}

// ------------- fp32 -> bf16 transposed weights  [mat][n][k] = w[mat][k][n] -------------
__global__ void k_wcvt(const float* __restrict__ w1, const float* __restrict__ w2,
                       uint16_t* __restrict__ w1t, uint16_t* __restrict__ w2t) {
  const int total = NLAYER * 2 * DIM * DIM;
  int t = blockIdx.x * blockDim.x + threadIdx.x;
  if (t >= 2 * total) return;
  const float* w = (t < total) ? w1 : w2;
  uint16_t*    o = (t < total) ? w1t : w2t;
  int i = (t < total) ? t : t - total;
  int mat = i / (DIM * DIM);
  int rem = i - mat * DIM * DIM;
  int n = rem >> 7, k = rem & 127;
  o[i] = f2bf(w[mat * DIM * DIM + k * DIM + n]);
}

// ---------------- edge scatter-add: agg[dst] += h[src] ----------------
__global__ void k_scatter(const int* __restrict__ src, const int* __restrict__ dst,
                          const float* __restrict__ h, float* agg) {
  int t = blockIdx.x * blockDim.x + threadIdx.x;      // NEDGES * 32 threads
  if (t >= NEDGES * 32) return;
  int e  = t >> 5;
  int c4 = (t & 31) << 2;
  int s = src[e], d = dst[e];
  float4 v = *(const float4*)(h + (size_t)s * DIM + c4);
  float* p = agg + (size_t)d * DIM + c4;
  __hip_atomic_fetch_add(p + 0, v.x, __ATOMIC_RELAXED, __HIP_MEMORY_SCOPE_AGENT);
  __hip_atomic_fetch_add(p + 1, v.y, __ATOMIC_RELAXED, __HIP_MEMORY_SCOPE_AGENT);
  __hip_atomic_fetch_add(p + 2, v.z, __ATOMIC_RELAXED, __HIP_MEMORY_SCOPE_AGENT);
  __hip_atomic_fetch_add(p + 3, v.w, __ATOMIC_RELAXED, __HIP_MEMORY_SCOPE_AGENT);
}

// ---------------- fused GIN MLP: out_r = relu((h+agg_r)@W1+b1)@W2+b2 ----------------
// One block = 128-node tile, one relation (blockIdx.y). Output aliases agg_r
// (block reads its own rows into LDS before overwriting them).
__global__ __launch_bounds__(256) void k_gin_mlp(
    const float* __restrict__ h, float* aggpos, float* agginv,
    const uint16_t* __restrict__ w1t, const float* __restrict__ b1b,
    const uint16_t* __restrict__ w2t, const float* __restrict__ b2b) {
  extern __shared__ uint16_t smem[];
  uint16_t* sA = smem;                 // 128 x 130 bf16 : A = h + agg
  uint16_t* sW = smem + DIM * LDU16;   // 128 x 130 bf16 : W^T (n-major), w1 then w2
  uint16_t* sZ = smem + 2 * DIM * LDU16; // 128 x 130 bf16 : relu intermediate
  uint32_t* sA32 = (uint32_t*)sA;
  uint32_t* sW32 = (uint32_t*)sW;
  uint32_t* sZ32 = (uint32_t*)sZ;

  const int r = blockIdx.y;
  float* agg = r ? agginv : aggpos;
  const uint16_t* w1 = w1t + r * DIM * DIM;
  const uint16_t* w2 = w2t + r * DIM * DIM;
  const float* b1 = b1b + r * DIM;
  const float* b2 = b2b + r * DIM;

  const int tile0 = blockIdx.x * 128;
  const int tid = threadIdx.x;

  // stage A (fp32 add + bf16 convert) and W1^T
  for (int t = tid; t < DIM * DIM; t += 256) {
    int row = t >> 7, col = t & 127;
    int node = tile0 + row;
    float v = 0.f;
    if (node < NNODES)
      v = h[(size_t)node * DIM + col] + agg[(size_t)node * DIM + col];
    sA[row * LDU16 + col] = f2bf(v);
    sW[row * LDU16 + col] = w1[t];     // already [n][k]
  }
  __syncthreads();

  const int wave = tid >> 5, lane = tid & 31;
  const int l15 = lane & 15, hsel = lane >> 4;   // A: K-half select; B: K-group select
  const int mbase = wave * 16;                    // this wave's 16-row strip

  v8f acc[8];
  #pragma unroll
  for (int i = 0; i < 8; ++i) acc[i] = (v8f){};

  // ---- GEMM1: Z = A @ W1 ----
  #pragma unroll
  for (int kk = 0; kk < 4; ++kk) {                // K0 = 32*kk
    const int K0d = kk * 16;                      // dword offset within row
    BF16x16 af;
    #pragma unroll
    for (int i = 0; i < 8; ++i) {
      int dw = K0d + 4 * hsel + (i < 4 ? i : i + 4);  // ISA A-layout (16-bit, 16x32)
      af.u[i] = sA32[(mbase + l15) * LDU32 + dw];
    }
    #pragma unroll
    for (int nt = 0; nt < 8; ++nt) {
      BF16x16 bf_;
      #pragma unroll
      for (int i = 0; i < 8; ++i)                  // ISA B-layout via transposed W
        bf_.u[i] = sW32[(nt * 16 + l15) * LDU32 + K0d + 4 * hsel + (i < 4 ? i : i + 4)];
      acc[nt] = __builtin_amdgcn_wmma_f32_16x16x32_bf16(
          false, af.v, false, bf_.v, (short)0, acc[nt], false, false);
    }
  }
  // bias + relu -> sZ (C/D layout: n = lane&15, m = vgpr + 8*(lane>>4))
  #pragma unroll
  for (int nt = 0; nt < 8; ++nt) {
    int n = nt * 16 + l15;
    float bb = b1[n];
    #pragma unroll
    for (int i = 0; i < 8; ++i) {
      int m = mbase + i + 8 * hsel;
      float v = acc[nt][i] + bb;
      sZ[m * LDU16 + n] = f2bf(v > 0.f ? v : 0.f);
    }
  }
  __syncthreads();
  // swap in W2^T
  for (int t = tid; t < DIM * DIM; t += 256)
    sW[(t >> 7) * LDU16 + (t & 127)] = w2[t];
  __syncthreads();

  #pragma unroll
  for (int i = 0; i < 8; ++i) acc[i] = (v8f){};

  // ---- GEMM2: OUT = Z @ W2 ----
  #pragma unroll
  for (int kk = 0; kk < 4; ++kk) {
    const int K0d = kk * 16;
    BF16x16 af;
    #pragma unroll
    for (int i = 0; i < 8; ++i) {
      int dw = K0d + 4 * hsel + (i < 4 ? i : i + 4);
      af.u[i] = sZ32[(mbase + l15) * LDU32 + dw];
    }
    #pragma unroll
    for (int nt = 0; nt < 8; ++nt) {
      BF16x16 bf_;
      #pragma unroll
      for (int i = 0; i < 8; ++i)
        bf_.u[i] = sW32[(nt * 16 + l15) * LDU32 + K0d + 4 * hsel + (i < 4 ? i : i + 4)];
      acc[nt] = __builtin_amdgcn_wmma_f32_16x16x32_bf16(
          false, af.v, false, bf_.v, (short)0, acc[nt], false, false);
    }
  }
  // bias + fp32 store (overwrites this block's rows of agg_r)
  #pragma unroll
  for (int nt = 0; nt < 8; ++nt) {
    int n = nt * 16 + l15;
    float bb = b2[n];
    #pragma unroll
    for (int i = 0; i < 8; ++i) {
      int m = mbase + i + 8 * hsel;
      int node = tile0 + m;
      if (node < NNODES) agg[(size_t)node * DIM + n] = acc[nt][i] + bb;
    }
  }
}

// ---------------- h = (relu?) (out_pos + out_inv) ----------------
__global__ void k_combine(float* __restrict__ h, const float* __restrict__ op,
                          const float* __restrict__ oi, int do_relu) {
  int t = blockIdx.x * blockDim.x + threadIdx.x;
  if (t >= NNODES * DIM) return;
  float v = op[t] + oi[t];
  h[t] = do_relu ? fmaxf(v, 0.f) : v;
}

// ---------------- pooling ----------------
__global__ void k_pool_init(unsigned* mx, unsigned* mn) {
  int t = blockIdx.x * blockDim.x + threadIdx.x;
  if (t >= NGRAPH * DIM) return;
  mx[t] = 0x007FFFFFu;   // flipf(-inf)
  mn[t] = 0xFF800000u;   // flipf(+inf)
}

__device__ __forceinline__ void pool_flush(int g, int c, float s, float vx, float vn,
                                           int run, float* sum, unsigned* mx,
                                           unsigned* mn, float* cnt) {
  __hip_atomic_fetch_add(&sum[g * DIM + c], s, __ATOMIC_RELAXED, __HIP_MEMORY_SCOPE_AGENT);
  atomicMax(&mx[g * DIM + c], flipf(vx));
  atomicMin(&mn[g * DIM + c], flipf(vn));
  if (c == 0)
    __hip_atomic_fetch_add(&cnt[g], (float)run, __ATOMIC_RELAXED, __HIP_MEMORY_SCOPE_AGENT);
}

// batch is sorted: accumulate runs in registers, atomically flush only on
// segment boundaries (~64 boundaries over 100k nodes -> few atomics).
__global__ void k_pool_scan(const float* __restrict__ h, const int* __restrict__ batch,
                            float* sum, unsigned* mx, unsigned* mn, float* cnt) {
  int c = threadIdx.x;                   // channel 0..127
  int n0 = blockIdx.x * 256;
  if (n0 >= NNODES) return;
  int n1 = n0 + 256; if (n1 > NNODES) n1 = NNODES;
  int curg = batch[n0];
  float s = 0.f, vx = -__builtin_inff(), vn = __builtin_inff();
  int run = 0;
  for (int n = n0; n < n1; ++n) {
    int g = batch[n];
    if (g != curg) {
      pool_flush(curg, c, s, vx, vn, run, sum, mx, mn, cnt);
      curg = g; s = 0.f; vx = -__builtin_inff(); vn = __builtin_inff(); run = 0;
    }
    float v = h[(size_t)n * DIM + c];
    s += v; vx = fmaxf(vx, v); vn = fminf(vn, v); ++run;
  }
  pool_flush(curg, c, s, vx, vn, run, sum, mx, mn, cnt);
}

// write torch-view interleaved output + batch_mask
__global__ void k_pool_final(const float* __restrict__ sum, const unsigned* __restrict__ mx,
                             const unsigned* __restrict__ mn, const float* __restrict__ cnt,
                             float* __restrict__ out) {
  int b = blockIdx.x, c = threadIdx.x;
  float s  = sum[b * DIM + c];
  float cn = cnt[b]; cn = cn > 1.f ? cn : 1.f;
  float me = s / cn;
  float vx = unflipf(mx[b * DIM + c]);
  float vn = unflipf(mn[b * DIM + c]);
  int j = c & 3, i = c >> 2;                       // concat k = s*128+c -> (j, i)
  float* ob = out + (size_t)(b * 4 + j) * DIM;
  ob[i]      = s;
  ob[32 + i] = me;
  ob[64 + i] = vx;
  ob[96 + i] = vn;
  if (c < 4) out[NGRAPH * 4 * DIM + b * 4 + c] = 1.0f;   // batch_mask = True
}

extern "C" void kernel_launch(void* const* d_in, const int* in_sizes, int n_in,
                              void* d_out, int out_size, void* d_ws, size_t ws_size,
                              hipStream_t stream) {
  const int*   x    = (const int*)d_in[0];
  const int*   eip  = (const int*)d_in[1];
  const int*   eii  = (const int*)d_in[2];
  const int*   batch= (const int*)d_in[3];
  const float* temb = (const float*)d_in[4];
  const float* oemb = (const float*)d_in[5];
  const float* l1w  = (const float*)d_in[6];
  const float* l1b  = (const float*)d_in[7];
  const float* l2w  = (const float*)d_in[8];
  const float* l2b  = (const float*)d_in[9];
  float* out = (float*)d_out;

  char* ws = (char*)d_ws;
  size_t off = 0;
  auto alloc = [&](size_t bytes) -> void* {
    void* p = ws + off;
    off = (off + bytes + 255) & ~(size_t)255;
    return p;
  };
  float*    h    = (float*)alloc(sizeof(float) * (size_t)NNODES * DIM);
  float*    agp  = (float*)alloc(sizeof(float) * (size_t)NNODES * DIM);
  float*    agi  = (float*)alloc(sizeof(float) * (size_t)NNODES * DIM);
  uint16_t* w1t  = (uint16_t*)alloc(sizeof(uint16_t) * NLAYER * 2 * DIM * DIM);
  uint16_t* w2t  = (uint16_t*)alloc(sizeof(uint16_t) * NLAYER * 2 * DIM * DIM);
  float*    gsum = (float*)alloc(sizeof(float) * NGRAPH * DIM);
  unsigned* gmx  = (unsigned*)alloc(sizeof(unsigned) * NGRAPH * DIM);
  unsigned* gmn  = (unsigned*)alloc(sizeof(unsigned) * NGRAPH * DIM);
  float*    gcnt = (float*)alloc(sizeof(float) * NGRAPH);

  dim3 b256(256);
  k_embed<<<(NNODES * DIM + 255) / 256, b256, 0, stream>>>(x, temb, oemb, h);
  k_wcvt<<<(2 * NLAYER * 2 * DIM * DIM + 255) / 256, b256, 0, stream>>>(l1w, l2w, w1t, w2t);

  const size_t shbytes = (size_t)3 * DIM * LDU16 * sizeof(uint16_t);  // ~100 KB / WG

  for (int l = 0; l < NLAYER; ++l) {
    hipMemsetAsync(agp, 0, sizeof(float) * (size_t)NNODES * DIM, stream);
    hipMemsetAsync(agi, 0, sizeof(float) * (size_t)NNODES * DIM, stream);
    int st = NEDGES * 32;
    k_scatter<<<(st + 255) / 256, b256, 0, stream>>>(eip, eip + NEDGES, h, agp);
    k_scatter<<<(st + 255) / 256, b256, 0, stream>>>(eii, eii + NEDGES, h, agi);
    dim3 grid((NNODES + 127) / 128, 2);
    k_gin_mlp<<<grid, b256, shbytes, stream>>>(h, agp, agi,
        w1t + l * 2 * DIM * DIM, l1b + l * 2 * DIM,
        w2t + l * 2 * DIM * DIM, l2b + l * 2 * DIM);
    k_combine<<<(NNODES * DIM + 255) / 256, b256, 0, stream>>>(
        h, agp, agi, (l < NLAYER - 1) ? 1 : 0);
  }

  hipMemsetAsync(gsum, 0, sizeof(float) * NGRAPH * DIM, stream);
  hipMemsetAsync(gcnt, 0, sizeof(float) * NGRAPH, stream);
  k_pool_init<<<(NGRAPH * DIM + 255) / 256, b256, 0, stream>>>(gmx, gmn);
  k_pool_scan<<<(NNODES + 255) / 256, dim3(128), 0, stream>>>(h, batch, gsum, gmx, gmn, gcnt);
  k_pool_final<<<NGRAPH, dim3(128), 0, stream>>>(gsum, gmx, gmn, gcnt, out);
}
